// TextRNNBinaryClassifier_74062416052734
// MI455X (gfx1250) — matrix operand order repaired and checked by reading, not compiled, not run on hardware
//
#include <hip/hip_runtime.h>
#include <hip/hip_bf16.h>

#define Bq 128
#define Sq 2048
#define Eq 128
#define Hq 64

typedef __attribute__((ext_vector_type(16))) _Float16 v16h;
typedef __attribute__((ext_vector_type(8)))  _Float16 v8h;
typedef __attribute__((ext_vector_type(8)))  float    v8f;
typedef __attribute__((ext_vector_type(4)))  unsigned int v4u;
typedef __attribute__((ext_vector_type(8)))  int v8i;
typedef __attribute__((ext_vector_type(4)))  int v4i;

union V16 { v16h v; v8h h[2]; _Float16 e[16]; };
union V8F { v8f v; float f[8]; };

__device__ __forceinline__ float ftanh(float x) {
#if __has_builtin(__builtin_amdgcn_tanhf)
    return __builtin_amdgcn_tanhf(x);
#elif __has_builtin(__builtin_amdgcn_tanh_f32)
    return __builtin_amdgcn_tanh_f32(x);
#else
    float e = __expf(2.0f * x);
    return 1.0f - 2.0f / (e + 1.0f);
#endif
}

// -------- f32 -> f16 weight conversion (once per launch) -------------
__global__ __launch_bounds__(256) void cvt_kernel(
    const float* __restrict__ src, _Float16* __restrict__ dst, int n)
{
    int i = blockIdx.x * 256 + threadIdx.x;
    if (i < n) dst[i] = (_Float16)src[i];
}

// -------- embedding gather -> f16 activations, TIME-MAJOR [s][b][e] --
__global__ __launch_bounds__(256) void gather_kernel(
    const int* __restrict__ x, const float* __restrict__ emb,
    _Float16* __restrict__ xe)
{
    long idx = (long)blockIdx.x * 256 + threadIdx.x;   // [0, S*B*E)
    long row = idx >> 7;            // s*B + b
    int  e   = (int)(idx & 127);
    int  s   = (int)(row >> 7);     // B == 128
    int  b   = (int)(row & 127);
    int tok = x[(long)b * Sq + s];
    xe[idx] = (_Float16)emb[(long)tok * Eq + e];
}

// ---------------- mean / max pooling over sequence -------------------
__global__ __launch_bounds__(128) void pool_kernel(
    const int* __restrict__ x, const float* __restrict__ emb,
    float* __restrict__ combined)   // [B,384]
{
    int b = blockIdx.x, e = threadIdx.x;
    float sum = 0.0f, mx = -__builtin_inff();
    for (int s = 0; s < Sq; ++s) {
        int tok = x[(long)b * Sq + s];
        float v = emb[(long)tok * Eq + e];
        sum += v; mx = fmaxf(mx, v);
    }
    combined[(long)b * 384 + 128 + e] = sum * (1.0f / Sq);
    combined[(long)b * 384 + 256 + e] = mx;
}

// -------- input projection GEMM: pre = in @ Wcat^T + (b_ih+b_hh) -----
// Time-major: one block == one time step (128 batch rows). The 32 KB
// A-slab is DMA'd into LDS once per block via the Tensor Data Mover
// (padded rows: 256B data + 16B pad to rotate LDS banks), then all 8
// waves serve A-fragments from LDS. B = f16 w_ih rows, direct global.
#define AROW 136   // 128 f16 data + 8 f16 pad per row

__global__ __launch_bounds__(256) void proj_kernel(
    const _Float16* __restrict__ in, const _Float16* __restrict__ wih16,
    const float* __restrict__ b_ih, const float* __restrict__ b_hh,
    float* __restrict__ pre, int layer)
{
    const int lane = threadIdx.x & 31;
    const int w    = threadIdx.x >> 5;      // wave = N-tile 0..7
    const int lo16 = lane & 15;
    const int hi   = lane >> 4;             // 0|1
    const long M0  = (long)blockIdx.x * 128;  // row = s*B + b, b = 0..127

    __align__(16) __shared__ _Float16 ain[128][AROW];

#if __has_builtin(__builtin_amdgcn_tensor_load_to_lds)
    if (threadIdx.x < 32) {
        unsigned      lds = (unsigned)(unsigned long)(&ain[0][0]);
        unsigned long ga  = (unsigned long)(in + M0 * Eq);
        // D# group0: count=1 | lds_addr | global_addr | type=2 (ISA 8.3)
        v4u g0 = { 1u, lds, (unsigned)ga,
                   (unsigned)(ga >> 32) | 0x80000000u };
        // D# group1 (ISA 8.4): data_size=1 (2B), pad_enable,
        // pad_interval=5 (64 DW = one 256B row), pad_amount=3 (4 DW = 16B),
        // tensor_dim0=128 @bit48, tensor_dim1=128 @bit80, tile_dim0=128
        // @bit112, tile_dim1=128 @bit128, tensor_dim0_stride=128 @bit160.
        v8i g1 = { (int)((1u << 16) | (1u << 20) | (5u << 22) | (3u << 25)),
                   (int)(128u << 16),
                   (int)(128u << 16),
                   (int)(128u << 16),
                   128,
                   128,
                   0, 0 };
        v4i g2 = { 0, 0, 0, 0 };
        v4i g3 = { 0, 0, 0, 0 };
        v8i g4 = { 0, 0, 0, 0, 0, 0, 0, 0 };
        __builtin_amdgcn_tensor_load_to_lds(g0, g1, g2, g3, g4, 0);
        __builtin_amdgcn_s_wait_tensorcnt(0);
    }
    __syncthreads();
#else
    // fallback: cooperative staged copy into the padded layout
    for (int i = threadIdx.x; i < 128 * 16; i += 256) {
        int row = i >> 4, col = (i & 15) * 8;
        *(v8h*)(&ain[row][col]) = *(const v8h*)(in + (M0 + row) * Eq + col);
    }
    __syncthreads();
#endif

    const int n = w * 16 + lo16;            // output column 0..127
    const int d = n >> 6;
    const int r = n & 63;
    const _Float16* wrow = wih16 + (((long)layer * 2 + d) * Hq + r) * Eq;
    const float biasn = b_ih[((long)layer * 2 + d) * Hq + r]
                      + b_hh[((long)layer * 2 + d) * Hq + r];

    V16 bfr[4];
    #pragma unroll
    for (int kt = 0; kt < 4; ++kt)
        bfr[kt].v = *(const v16h*)(wrow + kt * 32 + hi * 16);

    for (int mt = 0; mt < 8; ++mt) {
        const _Float16* ap = &ain[mt * 16 + lo16][0];
        V16 af[4];
        #pragma unroll
        for (int kt = 0; kt < 4; ++kt) {
            int kb = kt * 32 + hi * 16;
            af[kt].h[0] = *(const v8h*)(ap + kb);
            af[kt].h[1] = *(const v8h*)(ap + kb + 8);
        }
        V8F c;
        #pragma unroll
        for (int j = 0; j < 8; ++j) c.f[j] = biasn;
        #pragma unroll
        for (int kt = 0; kt < 4; ++kt)
            c.v = __builtin_amdgcn_wmma_f32_16x16x32_f16(
                false, af[kt].v, false, bfr[kt].v, (short)0, c.v, false, false);
        long mbase = M0 + mt * 16 + hi * 8;
        #pragma unroll
        for (int j = 0; j < 8; ++j)
            pre[(mbase + j) * 128 + n] = c.f[j];
    }
}

// -------- recurrent scan: h = tanh(pre + h @ W_hh^T), sequential S ----
// blockIdx.x = direction, blockIdx.y = batch band (16 rows). One wave,
// wave-private LDS state -> no barriers. Time-major pre; distance-2
// double-buffered prefetch of next steps' pre-activations.
__global__ __launch_bounds__(32) void scan_kernel(
    const float* __restrict__ pre, const _Float16* __restrict__ whh16,
    _Float16* __restrict__ outbuf, float* __restrict__ combined,
    int layer, int write_out, int write_final)
{
    const int d    = blockIdx.x;
    const int band = blockIdx.y;
    const int lane = threadIdx.x;
    const int lo16 = lane & 15;
    const int hi   = lane >> 4;
    const int m0   = band * 16;

    __shared__ _Float16 hbuf[16][72];   // padded rows, 16B-aligned stride
    for (int i = lane; i < 16 * 72; i += 32) ((_Float16*)hbuf)[i] = (_Float16)0;

    // B fragments: W_hh^T, B[k,n] = W[n,k] -> contiguous f16 W rows
    V16 bfr[4][2];
    #pragma unroll
    for (int nt = 0; nt < 4; ++nt) {
        int n = nt * 16 + lo16;
        const _Float16* Wr = whh16 + (((long)layer * 2 + d) * Hq + n) * Hq;
        #pragma unroll
        for (int kt = 0; kt < 2; ++kt)
            bfr[nt][kt].v = *(const v16h*)(Wr + kt * 32 + hi * 16);
    }

    // C-init gather for step s: contiguous [s][m0..m0+15][d*64 + n] slab
    auto load_cin = [&](int s, V8F dst[4]) {
        const float* base = pre + ((long)s * Bq + m0) * 128 + d * 64;
        #pragma unroll
        for (int nt = 0; nt < 4; ++nt) {
            int n = nt * 16 + lo16;
            #pragma unroll
            for (int j = 0; j < 8; ++j)
                dst[nt].f[j] = base[(long)(j + hi * 8) * 128 + n];
        }
    };
    auto step_s = [&](int it) { return (d == 0) ? it : (Sq - 1 - it); };

    V8F cinbuf[2][4];
    load_cin(step_s(0), cinbuf[0]);
    load_cin(step_s(1), cinbuf[1]);

    for (int it = 0; it < Sq; ++it) {
        int s = step_s(it);
        V8F* cin = cinbuf[it & 1];

        // A fragments from wave-private LDS h state
        V16 af[2];
        const _Float16* hp = &hbuf[lo16][0];
        #pragma unroll
        for (int kt = 0; kt < 2; ++kt) {
            int kb = kt * 32 + hi * 16;
            af[kt].h[0] = *(const v8h*)(hp + kb);
            af[kt].h[1] = *(const v8h*)(hp + kb + 8);
        }

        V8F acc[4];
        #pragma unroll
        for (int nt = 0; nt < 4; ++nt) {
            v8f c = cin[nt].v;
            c = __builtin_amdgcn_wmma_f32_16x16x32_f16(
                false, af[0].v, false, bfr[nt][0].v, (short)0, c, false, false);
            c = __builtin_amdgcn_wmma_f32_16x16x32_f16(
                false, af[1].v, false, bfr[nt][1].v, (short)0, c, false, false);
            acc[nt].v = c;
        }

        // pipeline: fetch pre for step it+2 while tanh/stores run
        if (it + 2 < Sq) load_cin(step_s(it + 2), cinbuf[it & 1]);

        #pragma unroll
        for (int nt = 0; nt < 4; ++nt) {
            int n = nt * 16 + lo16;
            #pragma unroll
            for (int j = 0; j < 8; ++j) {
                int mrow = j + hi * 8;
                float t = ftanh(acc[nt].f[j]);
                hbuf[mrow][n] = (_Float16)t;
                if (write_out) {
                    long row = (long)s * Bq + m0 + mrow;   // time-major out
                    outbuf[row * 128 + d * 64 + n] = (_Float16)t;
                }
                if (write_final && it == Sq - 1) {
                    long b = m0 + mrow;
                    combined[b * 384 + d * 64 + n] = t;
                }
            }
        }
    }
}

// ---------------- FC head: relu(combined@fc1^T+b1)@fc2^T+b2 ----------
__global__ __launch_bounds__(128) void fc_kernel(
    const float* __restrict__ combined, const float* __restrict__ fc1_w,
    const float* __restrict__ fc1_b, const float* __restrict__ fc2_w,
    const float* __restrict__ fc2_b, float* __restrict__ out)
{
    int b = blockIdx.x, j = threadIdx.x;
    const float* row = combined + (long)b * 384;
    float acc = fc1_b[j];
    for (int k = 0; k < 384; ++k) acc += row[k] * fc1_w[(long)j * 384 + k];
    float r = fmaxf(acc, 0.0f) * fc2_w[j];
    __shared__ float red[128];
    red[j] = r; __syncthreads();
    for (int st = 64; st > 0; st >>= 1) {
        if (j < st) red[j] += red[j + st];
        __syncthreads();
    }
    if (j == 0) out[b] = red[0] + fc2_b[0];
}

extern "C" void kernel_launch(void* const* d_in, const int* in_sizes, int n_in,
                              void* d_out, int out_size, void* d_ws, size_t ws_size,
                              hipStream_t stream) {
    const int*   x     = (const int*)  d_in[0];
    const float* emb   = (const float*)d_in[1];
    const float* w_ih  = (const float*)d_in[2];
    const float* w_hh  = (const float*)d_in[3];
    const float* b_ih  = (const float*)d_in[4];
    const float* b_hh  = (const float*)d_in[5];
    const float* fc1_w = (const float*)d_in[6];
    const float* fc1_b = (const float*)d_in[7];
    const float* fc2_w = (const float*)d_in[8];
    const float* fc2_b = (const float*)d_in[9];
    float* out = (float*)d_out;

    char* ws = (char*)d_ws;
    const size_t actBytes = (size_t)Bq * Sq * 128 * sizeof(_Float16); // 67 MB
    const size_t preBytes = (size_t)Bq * Sq * 128 * sizeof(float);    // 134 MB
    const int nWih = 3 * 2 * Hq * Eq;   // 49152
    const int nWhh = 3 * 2 * Hq * Hq;   // 24576
    _Float16* bufA     = (_Float16*)ws;
    _Float16* bufB     = (_Float16*)(ws + actBytes);
    float*    pre      = (float*)   (ws + 2 * actBytes);
    float*    combined = (float*)   (ws + 2 * actBytes + preBytes);
    _Float16* wih16    = (_Float16*)(ws + 2 * actBytes + preBytes + (size_t)Bq * 384 * 4);
    _Float16* whh16    = wih16 + nWih;

    // 0) one-time weight conversion to f16
    cvt_kernel<<<(nWih + 255) / 256, 256, 0, stream>>>(w_ih, wih16, nWih);
    cvt_kernel<<<(nWhh + 255) / 256, 256, 0, stream>>>(w_hh, whh16, nWhh);

    // 1) gather embeddings as f16 activations (time-major)
    gather_kernel<<<(Bq * Sq * Eq) / 256, 256, 0, stream>>>(x, emb, bufA);
    // 2) mean/max pooling -> combined[:,128:384]
    pool_kernel<<<Bq, 128, 0, stream>>>(x, emb, combined);

    // 3) three bidirectional RNN layers: projection GEMM + recurrent scan
    _Float16* bufs[2] = { bufA, bufB };
    for (int l = 0; l < 3; ++l) {
        _Float16* inb  = bufs[l & 1];
        _Float16* outb = bufs[(l + 1) & 1];
        proj_kernel<<<Sq, 256, 0, stream>>>(inb, wih16, b_ih, b_hh, pre, l);
        scan_kernel<<<dim3(2, 8), 32, 0, stream>>>(pre, whh16, outb, combined,
                                                   l, (l < 2) ? 1 : 0, (l == 2) ? 1 : 0);
    }

    // 4) FC head -> out[B,1]
    fc_kernel<<<Bq, 128, 0, stream>>>(combined, fc1_w, fc1_b, fc2_w, fc2_b, out);
}
